// myfix6_28784870818376
// MI455X (gfx1250) — compile-verified, matrix operand order checked by reading, not run on hardware
//
#include <hip/hip_runtime.h>
#include <math.h>

typedef __attribute__((ext_vector_type(16))) _Float16 v16h;
typedef __attribute__((ext_vector_type(8)))  float    v8f;

#define LPIX 4096
#define CCH  56
#define DICH 112
#define DSN  16
#define RR   4
#define KDIR 4
#define NBLK 10
#define NCHUNK 64
#define CLEN   64
#define NCHAN  (KDIR * DICH * DSN)   // 7168 scan channels

// ---- CDNA5 async global->LDS support (compile-safe detection) -------------
#if defined(__has_builtin)
#  if __has_builtin(__builtin_amdgcn_global_load_async_to_lds_b128)
#    define ASYNC_LDS 1
#  endif
#endif
#ifndef ASYNC_LDS
#  define ASYNC_LDS 0
#endif

#if ASYNC_LDS
typedef int v4i_async __attribute__((vector_size(16)));
#  if __has_builtin(__builtin_amdgcn_s_wait_asynccnt)
#    define WAIT_ASYNC() __builtin_amdgcn_s_wait_asynccnt(0)
#  else
#    define WAIT_ASYNC() asm volatile("s_wait_asynccnt 0x0" ::: "memory")
#  endif
#  define ASYNC_COPY_B128(dst_lds, src_glb)                                   \
     __builtin_amdgcn_global_load_async_to_lds_b128(                          \
         (__attribute__((address_space(1))) v4i_async*)(void*)(src_glb),      \
         (__attribute__((address_space(3))) v4i_async*)(dst_lds), 0, 0)
#endif

// ---------------------------------------------------------------------------
// WMMA GEMM, LDS-staged:  C[M,N] = A[M,K] * B[K,N]  (+bias[N]) (+resid[M,N])
// A addressed as A + row*a_rs + k*a_cs (rowmode 1: 64x64 pixel transpose of row)
// B addressed as B + k*b_sk + n*b_sn
// Block = 256 threads = 8 waves; block computes 128 rows x 16 cols.
// ---------------------------------------------------------------------------
#define SASTR 36   // elements per staged row (144B f32 rows / 72B f16 rows)

__global__ void gemm16(const float* __restrict__ A, int a_rs, int a_cs, int rowmode,
                       const float* __restrict__ B, int b_sk, int b_sn,
                       const float* __restrict__ bias,
                       const float* __restrict__ resid, int ld_res,
                       float* __restrict__ Cmat, int ldc,
                       int M, int N, int Kd)
{
#if ASYNC_LDS
    __shared__ __align__(16) float sAf[128 * SASTR];   // raw f32 tiles (async dest)
    __shared__ __align__(16) float sBf[16 * SASTR];
#else
    __shared__ __align__(16) _Float16 sA[128 * SASTR];
    __shared__ __align__(16) _Float16 sB[16 * SASTR];
#endif

    const int t     = threadIdx.x;
    const int lane  = t & 31;
    const int wave  = t >> 5;
    const int m0    = blockIdx.x * 128;
    const int ntile = blockIdx.y;
    const int khi   = (lane >= 16) ? 8 : 0;
    const bool vecA = (a_cs == 1);

    v8f acc = {};
    for (int k0 = 0; k0 < Kd; k0 += 32) {
        const bool fullK = (k0 + 32 <= Kd);

#if ASYNC_LDS
        // ============== async DMA staging of raw f32 tiles ==============
        const bool asyA = vecA && fullK;
        const bool asyB = (b_sk == 1) && fullK;
        if (asyA) {
#pragma unroll
            for (int i = 0; i < 4; ++i) {
                int idx = i * 256 + t;            // 1024 float4 slots
                int r   = idx >> 3;
                int q   = idx & 7;
                int gr  = m0 + r;
                if (rowmode == 1) gr = ((gr & 63) << 6) | (gr >> 6);
                ASYNC_COPY_B128(sAf + r * SASTR + q * 4,
                                A + (long)gr * a_rs + k0 + q * 4);
            }
        } else {
            float va[16];
#pragma unroll
            for (int i = 0; i < 16; ++i) {
                int idx = i * 256 + t;
                int r   = idx >> 5;
                int kk  = idx & 31;
                int kg  = k0 + kk;
                int kc  = (kg < Kd) ? kg : (Kd - 1);
                int gr  = m0 + r;
                if (rowmode == 1) gr = ((gr & 63) << 6) | (gr >> 6);
                va[i] = A[(long)gr * a_rs + (long)kc * a_cs];
            }
#pragma unroll
            for (int i = 0; i < 16; ++i) {
                int idx = i * 256 + t;
                int r   = idx >> 5;
                int kk  = idx & 31;
                sAf[r * SASTR + kk] = (k0 + kk < Kd) ? va[i] : 0.f;
            }
        }
        if (asyB) {
            if (t < 128) {
                int n  = t >> 3;
                int q  = t & 7;
                int ng = ntile * 16 + n;
                int nc = (ng < N) ? ng : (N - 1);   // clamped cols: outputs never stored
                ASYNC_COPY_B128(sBf + n * SASTR + q * 4,
                                B + (long)nc * b_sn + k0 + q * 4);
            }
        } else {
            float vb[2];
#pragma unroll
            for (int i = 0; i < 2; ++i) {
                int idx = i * 256 + t;
                int kk  = idx >> 4;
                int n   = idx & 15;
                int kg  = k0 + kk;
                int ng  = ntile * 16 + n;
                int kc  = (kg < Kd) ? kg : (Kd - 1);
                int nc  = (ng < N) ? ng : (N - 1);
                vb[i] = B[(long)kc * b_sk + (long)nc * b_sn];
            }
#pragma unroll
            for (int i = 0; i < 2; ++i) {
                int idx = i * 256 + t;
                int kk  = idx >> 4;
                int n   = idx & 15;
                sBf[n * SASTR + kk] = (k0 + kk < Kd) ? vb[i] : 0.f;
            }
        }
        if (asyA || asyB) WAIT_ASYNC();
        __syncthreads();

        // ---- fragments: f32 pairs from LDS -> f16 convert ----
        v16h ah, bh;
        {
            const float2* pa2 = (const float2*)(sAf + (size_t)(wave * 16 + (lane & 15)) * SASTR);
            const float2* pb2 = (const float2*)(sBf + (size_t)(lane & 15) * SASTR);
            int kb = khi >> 1;                    // 0 or 4 float2 slots
#pragma unroll
            for (int i = 0; i < 4; ++i) {
                float2 a0 = pa2[kb + i];          // halves khi+2i, khi+2i+1
                float2 a1 = pa2[kb + 8 + i];      // halves khi+16+2i, ...
                float2 b0 = pb2[kb + i];
                float2 b1 = pb2[kb + 8 + i];
                ah[2 * i]     = (_Float16)a0.x;  ah[2 * i + 1] = (_Float16)a0.y;
                ah[8 + 2 * i] = (_Float16)a1.x;  ah[9 + 2 * i] = (_Float16)a1.y;
                bh[2 * i]     = (_Float16)b0.x;  bh[2 * i + 1] = (_Float16)b0.y;
                bh[8 + 2 * i] = (_Float16)b1.x;  bh[9 + 2 * i] = (_Float16)b1.y;
            }
        }
        acc = __builtin_amdgcn_wmma_f32_16x16x32_f16(false, ah, false, bh,
                                                     (short)0, acc, false, false);
        __syncthreads();
#else
        // ============== synchronous staging (f16 in LDS) ==============
        if (vecA) {
            float4 va[4];
#pragma unroll
            for (int i = 0; i < 4; ++i) {
                int idx = i * 256 + t;
                int r   = idx >> 3;
                int q   = idx & 7;
                int gr  = m0 + r;
                if (rowmode == 1) gr = ((gr & 63) << 6) | (gr >> 6);
                const float* ap = A + (long)gr * a_rs + k0 + q * 4;
                if (k0 + q * 4 + 3 < Kd) {
                    va[i] = *reinterpret_cast<const float4*>(ap);
                } else {
                    float e[4];
#pragma unroll
                    for (int j = 0; j < 4; ++j) {
                        int kg = k0 + q * 4 + j;
                        int kc = (kg < Kd) ? kg : (Kd - 1);
                        e[j] = A[(long)gr * a_rs + kc];
                    }
                    va[i] = make_float4(e[0], e[1], e[2], e[3]);
                }
            }
#pragma unroll
            for (int i = 0; i < 4; ++i) {
                int idx = i * 256 + t;
                int r   = idx >> 3;
                int q   = idx & 7;
                int kg0 = k0 + q * 4;
                union { _Float16 h[4]; uint2 u; } pk;
                pk.h[0] = (kg0 + 0 < Kd) ? (_Float16)va[i].x : (_Float16)0.f;
                pk.h[1] = (kg0 + 1 < Kd) ? (_Float16)va[i].y : (_Float16)0.f;
                pk.h[2] = (kg0 + 2 < Kd) ? (_Float16)va[i].z : (_Float16)0.f;
                pk.h[3] = (kg0 + 3 < Kd) ? (_Float16)va[i].w : (_Float16)0.f;
                *reinterpret_cast<uint2*>(sA + r * SASTR + q * 4) = pk.u;
            }
        } else {
            float va[16];
#pragma unroll
            for (int i = 0; i < 16; ++i) {
                int idx = i * 256 + t;
                int r   = idx >> 5;
                int kk  = idx & 31;
                int kg  = k0 + kk;
                int kc  = (kg < Kd) ? kg : (Kd - 1);
                int gr  = m0 + r;
                if (rowmode == 1) gr = ((gr & 63) << 6) | (gr >> 6);
                va[i] = A[(long)gr * a_rs + (long)kc * a_cs];
            }
#pragma unroll
            for (int i = 0; i < 16; ++i) {
                int idx = i * 256 + t;
                int r   = idx >> 5;
                int kk  = idx & 31;
                sA[r * SASTR + kk] = (k0 + kk < Kd) ? (_Float16)va[i] : (_Float16)0.f;
            }
        }
        if (b_sk == 1) {
            if (t < 128) {
                int n  = t >> 3;
                int q  = t & 7;
                int ng = ntile * 16 + n;
                int nc = (ng < N) ? ng : (N - 1);
                float4 vb;
                if (k0 + q * 4 + 3 < Kd) {
                    vb = *reinterpret_cast<const float4*>(B + (long)nc * b_sn + k0 + q * 4);
                } else {
                    float e[4];
#pragma unroll
                    for (int j = 0; j < 4; ++j) {
                        int kg = k0 + q * 4 + j;
                        int kc = (kg < Kd) ? kg : (Kd - 1);
                        e[j] = B[(long)nc * b_sn + kc];
                    }
                    vb = make_float4(e[0], e[1], e[2], e[3]);
                }
                int  kg0 = k0 + q * 4;
                bool nok = (ng < N);
                union { _Float16 h[4]; uint2 u; } pk;
                pk.h[0] = (nok && kg0 + 0 < Kd) ? (_Float16)vb.x : (_Float16)0.f;
                pk.h[1] = (nok && kg0 + 1 < Kd) ? (_Float16)vb.y : (_Float16)0.f;
                pk.h[2] = (nok && kg0 + 2 < Kd) ? (_Float16)vb.z : (_Float16)0.f;
                pk.h[3] = (nok && kg0 + 3 < Kd) ? (_Float16)vb.w : (_Float16)0.f;
                *reinterpret_cast<uint2*>(sB + n * SASTR + q * 4) = pk.u;
            }
        } else {
            float vb[2];
#pragma unroll
            for (int i = 0; i < 2; ++i) {
                int idx = i * 256 + t;
                int kk  = idx >> 4;
                int n   = idx & 15;
                int kg  = k0 + kk;
                int ng  = ntile * 16 + n;
                int kc  = (kg < Kd) ? kg : (Kd - 1);
                int nc  = (ng < N) ? ng : (N - 1);
                vb[i] = B[(long)kc * b_sk + (long)nc * b_sn];
            }
#pragma unroll
            for (int i = 0; i < 2; ++i) {
                int idx = i * 256 + t;
                int kk  = idx >> 4;
                int n   = idx & 15;
                sB[n * SASTR + kk] =
                    (k0 + kk < Kd && ntile * 16 + n < N) ? (_Float16)vb[i] : (_Float16)0.f;
            }
        }
        if (vecA && fullK && k0 + 32 < Kd)
            __builtin_prefetch(A + (long)(m0 + (t >> 1)) * a_rs + k0 + 32, 0, 1);
        __syncthreads();

        union { v16h h; unsigned u[8]; } afu, bfu;
        {
            const unsigned* pa =
                (const unsigned*)(sA + (size_t)(wave * 16 + (lane & 15)) * SASTR);
            const unsigned* pb =
                (const unsigned*)(sB + (size_t)(lane & 15) * SASTR);
            int kb = khi >> 1;
#pragma unroll
            for (int i = 0; i < 4; ++i) {
                afu.u[i]     = pa[kb + i];
                afu.u[4 + i] = pa[kb + 8 + i];
                bfu.u[i]     = pb[kb + i];
                bfu.u[4 + i] = pb[kb + 8 + i];
            }
        }
        acc = __builtin_amdgcn_wmma_f32_16x16x32_f16(false, afu.h, false, bfu.h,
                                                     (short)0, acc, false, false);
        __syncthreads();
#endif
    }

    int nout = ntile * 16 + (lane & 15);
    if (nout < N) {
        float bv = bias ? bias[nout] : 0.f;
#pragma unroll
        for (int i = 0; i < 8; ++i) {
            int mo = m0 + wave * 16 + i + ((lane >= 16) ? 8 : 0);
            float v = acc[i] + bv;
            if (resid) v += resid[(long)mo * ld_res + nout];
            Cmat[(long)mo * ldc + nout] = v;
        }
    }
}

// ---------------------------------------------------------------------------
// LayerNorm over channel dim (channels-last, C<=64), optional exact GELU.
// ---------------------------------------------------------------------------
__global__ void ln_ch(const float* __restrict__ x, int ldx,
                      const float* __restrict__ w, const float* __restrict__ b,
                      float eps, int C, int dogelu,
                      float* __restrict__ out, int ldo)
{
    __shared__ float sh[64];
    int p = blockIdx.x, t = threadIdx.x;
    float v = (t < C) ? x[(long)p * ldx + t] : 0.f;
    sh[t] = v; __syncthreads();
    for (int s = 32; s > 0; s >>= 1) { if (t < s) sh[t] += sh[t + s]; __syncthreads(); }
    float mu = sh[0] / (float)C; __syncthreads();
    float dv = (t < C) ? (v - mu) : 0.f;
    sh[t] = dv * dv; __syncthreads();
    for (int s = 32; s > 0; s >>= 1) { if (t < s) sh[t] += sh[t + s]; __syncthreads(); }
    float var = sh[0] / (float)C;
    if (t < C) {
        float y = dv * rsqrtf(var + eps) * w[t] + b[t];
        if (dogelu) y = 0.5f * y * (1.f + erff(y * 0.70710678118f));
        out[(long)p * ldo + t] = y;
    }
}

// ---------------------------------------------------------------------------
// Depthwise 3x3, channels-last, pad=1; optional SiLU, optional residual add.
// ---------------------------------------------------------------------------
__global__ void dwconv(const float* __restrict__ in, int ldin, int inoff,
                       const float* __restrict__ w9, const float* __restrict__ bias,
                       const float* __restrict__ res, int ldres,
                       float* __restrict__ out, int ldo, int outoff,
                       int C, int act)
{
    int idx = blockIdx.x * blockDim.x + threadIdx.x;
    if (idx >= LPIX * C) return;
    int c = idx % C, p = idx / C;
    int h = p >> 6, w = p & 63;
    float acc = bias ? bias[c] : 0.f;
#pragma unroll
    for (int ty = 0; ty < 3; ++ty)
#pragma unroll
        for (int tx = 0; tx < 3; ++tx) {
            int hh = h + ty - 1, ww = w + tx - 1;
            if (hh >= 0 && hh < 64 && ww >= 0 && ww < 64)
                acc += w9[c * 9 + ty * 3 + tx] * in[(long)(hh * 64 + ww) * ldin + inoff + c];
        }
    if (act == 1) acc = acc / (1.f + expf(-acc));       // SiLU
    if (res) acc += res[(long)p * ldres + c];
    out[(long)p * ldo + outoff + c] = acc;
}

// ---------------------------------------------------------------------------
// Chunked selective scan (pass1 local states / prefix stitch / pass2 replay).
// ---------------------------------------------------------------------------
__global__ void scan_pass1(const float* __restrict__ G,
                           const float* __restrict__ XC,
                           const float* __restrict__ dtw, const float* __restrict__ dtb,
                           const float* __restrict__ Alog,
                           float* __restrict__ Hend, float* __restrict__ Pprod)
{
    int cidx = blockIdx.x / 28;
    int sub  = blockIdx.x % 28;
    int k = sub / 7;
    int d = (sub % 7) * 16 + (threadIdx.x >> 4);
    int n = threadIdx.x & 15;

    const float* Gk = G + (long)k * LPIX * 36;
    float w0 = dtw[(k * DICH + d) * 4 + 0];
    float w1 = dtw[(k * DICH + d) * 4 + 1];
    float w2 = dtw[(k * DICH + d) * 4 + 2];
    float w3 = dtw[(k * DICH + d) * 4 + 3];
    float bdt  = dtb[k * DICH + d];
    float Areg = -expf(Alog[(long)(k * DICH + d) * DSN + n]);

    float hloc = 0.f, pl = 1.f;
    int t0 = cidx * CLEN;
    for (int tt = 0; tt < CLEN; ++tt) {
        int tq  = t0 + tt;
        int g   = (k < 2) ? tq : (LPIX - 1 - tq);
        int pix = (k & 1) ? (((g & 63) << 6) | (g >> 6)) : g;
        const float* gr = Gk + (long)g * 36;
        float dts   = gr[0] * w0 + gr[1] * w1 + gr[2] * w2 + gr[3] * w3 + bdt;
        float delta = (dts > 20.f) ? dts : log1pf(expf(dts));
        float xv = XC[(long)pix * DICH + d];
        float a  = expf(delta * Areg);
        hloc = a * hloc + (delta * xv) * gr[RR + n];
        pl  *= a;
    }
    int ch = (k * DICH + d) * DSN + n;
    Hend [(long)cidx * NCHAN + ch] = hloc;
    Pprod[(long)cidx * NCHAN + ch] = pl;
}

__global__ void scan_prefix(const float* __restrict__ Hend,
                            const float* __restrict__ Pprod,
                            float* __restrict__ H0)
{
    int ch = blockIdx.x * blockDim.x + threadIdx.x;
    if (ch >= NCHAN) return;
    float h = 0.f;
    for (int c = 0; c < NCHUNK; ++c) {
        H0[(long)c * NCHAN + ch] = h;
        h = Pprod[(long)c * NCHAN + ch] * h + Hend[(long)c * NCHAN + ch];
    }
}

__global__ void scan_pass2(const float* __restrict__ G,
                           const float* __restrict__ XC,
                           const float* __restrict__ dtw, const float* __restrict__ dtb,
                           const float* __restrict__ Alog, const float* __restrict__ Dp,
                           const float* __restrict__ H0,
                           float* __restrict__ Yout)
{
    int cidx = blockIdx.x / 28;
    int sub  = blockIdx.x % 28;
    int k = sub / 7;
    int d = (sub % 7) * 16 + (threadIdx.x >> 4);
    int n = threadIdx.x & 15;

    const float* Gk = G + (long)k * LPIX * 36;
    float w0 = dtw[(k * DICH + d) * 4 + 0];
    float w1 = dtw[(k * DICH + d) * 4 + 1];
    float w2 = dtw[(k * DICH + d) * 4 + 2];
    float w3 = dtw[(k * DICH + d) * 4 + 3];
    float bdt  = dtb[k * DICH + d];
    float Areg = -expf(Alog[(long)(k * DICH + d) * DSN + n]);
    float Dreg = Dp[k * DICH + d];
    float* Yk  = Yout + (long)k * LPIX * DICH;

    int ch = (k * DICH + d) * DSN + n;
    float hstate = H0[(long)cidx * NCHAN + ch];

    int t0 = cidx * CLEN;
    for (int tt = 0; tt < CLEN; ++tt) {
        int tq  = t0 + tt;
        int g   = (k < 2) ? tq : (LPIX - 1 - tq);
        int pix = (k & 1) ? (((g & 63) << 6) | (g >> 6)) : g;
        const float* gr = Gk + (long)g * 36;
        float dts   = gr[0] * w0 + gr[1] * w1 + gr[2] * w2 + gr[3] * w3 + bdt;
        float delta = (dts > 20.f) ? dts : log1pf(expf(dts));
        float xv = XC[(long)pix * DICH + d];
        hstate = expf(delta * Areg) * hstate + (delta * xv) * gr[RR + n];
        float part = hstate * gr[RR + DSN + n];
        part += __shfl_xor(part, 8, 16);
        part += __shfl_xor(part, 4, 16);
        part += __shfl_xor(part, 2, 16);
        part += __shfl_xor(part, 1, 16);
        if (n == 0) Yk[(long)tq * DICH + d] = part + Dreg * xv;
    }
}

// ---------------------------------------------------------------------------
// Combine 4 scan directions (index remap), LN over DI=112, SiLU(z) gate.
// ---------------------------------------------------------------------------
__global__ void combine_ln_gate(const float* __restrict__ Y,
                                const float* __restrict__ XZ,
                                const float* __restrict__ ow, const float* __restrict__ ob,
                                float* __restrict__ out)
{
    __shared__ float sh[128];
    int p = blockIdx.x, t = threadIdx.x;
    int h = p >> 6, w = p & 63;
    int pt = w * 64 + h;
    float v = 0.f;
    if (t < DICH) {
        v = Y[(long)p * DICH + t]
          + Y[(long)(1 * LPIX + pt) * DICH + t]
          + Y[(long)(2 * LPIX + (LPIX - 1 - p)) * DICH + t]
          + Y[(long)(3 * LPIX + (LPIX - 1 - pt)) * DICH + t];
    }
    sh[t] = (t < DICH) ? v : 0.f; __syncthreads();
    for (int s = 64; s > 0; s >>= 1) { if (t < s) sh[t] += sh[t + s]; __syncthreads(); }
    float mu = sh[0] / (float)DICH; __syncthreads();
    float dv = (t < DICH) ? (v - mu) : 0.f;
    sh[t] = dv * dv; __syncthreads();
    for (int s = 64; s > 0; s >>= 1) { if (t < s) sh[t] += sh[t + s]; __syncthreads(); }
    float var = sh[0] / (float)DICH;
    if (t < DICH) {
        float y = dv * rsqrtf(var + 1e-5f) * ow[t] + ob[t];
        float z = XZ[(long)p * 224 + DICH + t];
        out[(long)p * DICH + t] = y * (z / (1.f + expf(-z)));
    }
}

// ---------------------------------------------------------------------------
__global__ void maxpool14(const float* __restrict__ xln, int goff, int f, int S,
                          float* __restrict__ pooled)
{
    int idx = blockIdx.x * blockDim.x + threadIdx.x;
    if (idx >= S * S * 14) return;
    int c = idx % 14, pp = idx / 14;
    int ph = pp / S, pw = pp % S;
    float m = -3.4e38f;
    for (int dy = 0; dy < f; ++dy)
        for (int dx = 0; dx < f; ++dx)
            m = fmaxf(m, xln[(long)((ph * f + dy) * 64 + pw * f + dx) * CCH + goff + c]);
    pooled[(long)pp * 14 + c] = m;
}

__global__ void dwup14(const float* __restrict__ pooled, int S, int f,
                       const float* __restrict__ w9, const float* __restrict__ bias,
                       float* __restrict__ out, int outoff)
{
    int idx = blockIdx.x * blockDim.x + threadIdx.x;
    if (idx >= LPIX * 14) return;
    int c = idx % 14, p = idx / 14;
    int ph = (p >> 6) / f, pw = (p & 63) / f;
    float acc = bias[c];
#pragma unroll
    for (int ty = 0; ty < 3; ++ty)
#pragma unroll
        for (int tx = 0; tx < 3; ++tx) {
            int hh = ph + ty - 1, ww = pw + tx - 1;
            if (hh >= 0 && hh < S && ww >= 0 && ww < S)
                acc += w9[c * 9 + ty * 3 + tx] * pooled[(long)(hh * S + ww) * 14 + c];
        }
    out[(long)p * CCH + outoff + c] = acc;
}

__global__ void gate_res(const float* __restrict__ XN, const float* __restrict__ AG,
                         const float* __restrict__ XLN,
                         float* __restrict__ XCUR, float* __restrict__ trunk, int nb)
{
    int idx = blockIdx.x * blockDim.x + threadIdx.x;
    if (idx >= LPIX * CCH) return;
    int c = idx % CCH, p = idx / CCH;
    float a = AG[idx];
    float g = 0.5f * a * (1.f + erff(a * 0.70710678118f));
    float v = XN[idx] + g * XLN[idx];
    XCUR[idx] = v;
    trunk[(long)p * (NBLK * CCH) + nb * CCH + c] = v;
}

__global__ void upshuffle(const float* __restrict__ outlr,
                          const float* __restrict__ upw, const float* __restrict__ upb,
                          float* __restrict__ out)
{
    int idx = blockIdx.x * blockDim.x + threadIdx.x;
    if (idx >= LPIX * 27) return;
    int u = idx % 27, p = idx / 27;
    int h = p >> 6, w = p & 63;
    float acc = upb[u];
    for (int ty = 0; ty < 3; ++ty)
        for (int tx = 0; tx < 3; ++tx) {
            int hh = h + ty - 1, ww = w + tx - 1;
            if (hh < 0 || hh >= 64 || ww < 0 || ww >= 64) continue;
            const float* row = outlr + (long)(hh * 64 + ww) * CCH;
            const float* wk  = upw + ((long)u * CCH) * 9 + ty * 3 + tx;
            for (int ic = 0; ic < CCH; ++ic)
                acc += wk[ic * 9] * row[ic];
        }
    int c = u / 9, i = (u % 9) / 3, j = u % 3;
    out[(long)c * 192 * 192 + (long)(h * 3 + i) * 192 + (w * 3 + j)] = acc;
}

// ---------------------------------------------------------------------------
extern "C" void kernel_launch(void* const* d_in, const int* in_sizes, int n_in,
                              void* d_out, int out_size, void* d_ws, size_t ws_size,
                              hipStream_t stream)
{
    (void)in_sizes; (void)n_in; (void)out_size; (void)ws_size;
    const float* x          = (const float*)d_in[0];
    const float* fea_pw_w   = (const float*)d_in[1];
    const float* fea_dw_w   = (const float*)d_in[2];
    const float* fea_dw_b   = (const float*)d_in[3];
    const float* b_ln1_w    = (const float*)d_in[4];
    const float* b_ln1_b    = (const float*)d_in[5];
    const float* b_inproj   = (const float*)d_in[6];
    const float* b_conv_w   = (const float*)d_in[7];
    const float* b_conv_b   = (const float*)d_in[8];
    const float* b_xproj    = (const float*)d_in[9];
    const float* b_dt_w     = (const float*)d_in[10];
    const float* b_dt_b     = (const float*)d_in[11];
    const float* b_Alog     = (const float*)d_in[12];
    const float* b_D        = (const float*)d_in[13];
    const float* b_onorm_w  = (const float*)d_in[14];
    const float* b_onorm_b  = (const float*)d_in[15];
    const float* b_outproj  = (const float*)d_in[16];
    const float* b_norm_w   = (const float*)d_in[17];
    const float* b_norm_b   = (const float*)d_in[18];
    const float* b_mfr_w    = (const float*)d_in[19];
    const float* b_mfr_b    = (const float*)d_in[20];
    const float* b_aggr_w   = (const float*)d_in[21];
    const float* b_aggr_b   = (const float*)d_in[22];
    const float* c1_w       = (const float*)d_in[23];
    const float* c1_b       = (const float*)d_in[24];
    const float* norm_w     = (const float*)d_in[25];
    const float* norm_b     = (const float*)d_in[26];
    const float* c2_pw_w    = (const float*)d_in[27];
    const float* c2_dw_w    = (const float*)d_in[28];
    const float* c2_dw_b    = (const float*)d_in[29];
    const float* up_w       = (const float*)d_in[30];
    const float* up_b       = (const float*)d_in[31];

    float* ws = (float*)d_ws;
    size_t o = 0;
    float* FEA    = ws + o; o += (size_t)LPIX * CCH;
    float* XCUR   = ws + o; o += (size_t)LPIX * CCH;
    float* XN     = ws + o; o += (size_t)LPIX * CCH;
    float* XLN    = ws + o; o += (size_t)LPIX * CCH;
    float* XZ     = ws + o; o += (size_t)LPIX * 224;
    float* XC     = ws + o; o += (size_t)LPIX * DICH;
    float* GB     = ws + o; o += (size_t)4 * LPIX * 36;
    float* YB     = ws + o; o += (size_t)4 * LPIX * DICH;
    float* GATED  = ws + o; o += (size_t)LPIX * DICH;
    float* CONC   = ws + o; o += (size_t)LPIX * CCH;
    float* POOL1  = ws + o; o += (size_t)32 * 32 * 14;
    float* POOL2  = ws + o; o += (size_t)16 * 16 * 14;
    float* POOL3  = ws + o; o += (size_t)8 * 8 * 14;
    float* AG     = ws + o; o += (size_t)LPIX * CCH;
    float* TRUNK  = ws + o; o += (size_t)LPIX * NBLK * CCH;
    float* OUTB   = ws + o; o += (size_t)LPIX * CCH;
    float* OUTLR  = ws + o; o += (size_t)LPIX * CCH;
    float* HEND   = ws + o; o += (size_t)NCHUNK * NCHAN;
    float* PPROD  = ws + o; o += (size_t)NCHUNK * NCHAN;
    float* H0B    = ws + o; o += (size_t)NCHUNK * NCHAN;

    const int EL = 256;

    gemm16<<<dim3(32, 4), 256, 0, stream>>>(x, 1, LPIX, 0,
                                            fea_pw_w, 1, 3, nullptr, nullptr, 0,
                                            XN, CCH, LPIX, CCH, 3);
    dwconv<<<(LPIX * CCH + EL - 1) / EL, EL, 0, stream>>>(XN, CCH, 0, fea_dw_w, fea_dw_b,
                                                          nullptr, 0, FEA, CCH, 0, CCH, 0);
    (void)hipMemcpyAsync(XCUR, FEA, sizeof(float) * LPIX * CCH,
                         hipMemcpyDeviceToDevice, stream);

    for (int nb = 0; nb < NBLK; ++nb) {
        ln_ch<<<LPIX, 64, 0, stream>>>(XCUR, CCH, b_ln1_w + nb * CCH, b_ln1_b + nb * CCH,
                                       1e-5f, CCH, 0, XLN, CCH);
        gemm16<<<dim3(32, 14), 256, 0, stream>>>(XLN, CCH, 1, 0,
                                                 b_inproj + (long)nb * 224 * CCH, 1, CCH,
                                                 nullptr, nullptr, 0,
                                                 XZ, 224, LPIX, 224, CCH);
        dwconv<<<(LPIX * DICH + EL - 1) / EL, EL, 0, stream>>>(XZ, 224, 0,
                                                               b_conv_w + nb * DICH * 9,
                                                               b_conv_b + nb * DICH,
                                                               nullptr, 0, XC, DICH, 0, DICH, 1);
        for (int k = 0; k < KDIR; ++k)
            gemm16<<<dim3(32, 3), 256, 0, stream>>>(XC, DICH, 1, (k & 1),
                                                    b_xproj + (long)nb * 4 * 36 * DICH + (long)k * 36 * DICH,
                                                    1, DICH, nullptr, nullptr, 0,
                                                    GB + (long)k * LPIX * 36, 36, LPIX, 36, DICH);
        scan_pass1<<<NCHUNK * 28, 256, 0, stream>>>(GB, XC,
                                                    b_dt_w + (long)nb * KDIR * DICH * RR,
                                                    b_dt_b + (long)nb * KDIR * DICH,
                                                    b_Alog + (long)nb * KDIR * DICH * DSN,
                                                    HEND, PPROD);
        scan_prefix<<<(NCHAN + EL - 1) / EL, EL, 0, stream>>>(HEND, PPROD, H0B);
        scan_pass2<<<NCHUNK * 28, 256, 0, stream>>>(GB, XC,
                                                    b_dt_w + (long)nb * KDIR * DICH * RR,
                                                    b_dt_b + (long)nb * KDIR * DICH,
                                                    b_Alog + (long)nb * KDIR * DICH * DSN,
                                                    b_D + (long)nb * KDIR * DICH,
                                                    H0B, YB);
        combine_ln_gate<<<LPIX, 128, 0, stream>>>(YB, XZ,
                                                  b_onorm_w + nb * DICH, b_onorm_b + nb * DICH,
                                                  GATED);
        gemm16<<<dim3(32, 4), 256, 0, stream>>>(GATED, DICH, 1, 0,
                                                b_outproj + (long)nb * CCH * DICH, 1, DICH,
                                                nullptr, XCUR, CCH,
                                                XN, CCH, LPIX, CCH, DICH);
        ln_ch<<<LPIX, 64, 0, stream>>>(XN, CCH, b_norm_w + nb * CCH, b_norm_b + nb * CCH,
                                       1e-6f, CCH, 0, XLN, CCH);
        dwconv<<<(LPIX * 14 + EL - 1) / EL, EL, 0, stream>>>(XLN, CCH, 0,
                                                             b_mfr_w + nb * 504,
                                                             b_mfr_b + nb * 56,
                                                             nullptr, 0, CONC, CCH, 0, 14, 0);
        float* pools[3] = {POOL1, POOL2, POOL3};
        for (int i = 1; i < 4; ++i) {
            int f = 1 << i, S = 64 >> i;
            maxpool14<<<(S * S * 14 + EL - 1) / EL, EL, 0, stream>>>(XLN, i * 14, f, S, pools[i - 1]);
            dwup14<<<(LPIX * 14 + EL - 1) / EL, EL, 0, stream>>>(pools[i - 1], S, f,
                                                                 b_mfr_w + nb * 504 + i * 126,
                                                                 b_mfr_b + nb * 56 + i * 14,
                                                                 CONC, i * 14);
        }
        gemm16<<<dim3(32, 4), 256, 0, stream>>>(CONC, CCH, 1, 0,
                                                b_aggr_w + (long)nb * CCH * CCH, 1, CCH,
                                                b_aggr_b + nb * CCH, nullptr, 0,
                                                AG, CCH, LPIX, CCH, CCH);
        gate_res<<<(LPIX * CCH + EL - 1) / EL, EL, 0, stream>>>(XN, AG, XLN, XCUR, TRUNK, nb);
    }

    gemm16<<<dim3(32, 4), 256, 0, stream>>>(TRUNK, NBLK * CCH, 1, 0,
                                            c1_w, 1, NBLK * CCH, c1_b, nullptr, 0,
                                            OUTB, CCH, LPIX, CCH, NBLK * CCH);
    ln_ch<<<LPIX, 64, 0, stream>>>(OUTB, CCH, norm_w, norm_b, 1e-6f, CCH, 1, XLN, CCH);
    gemm16<<<dim3(32, 4), 256, 0, stream>>>(XLN, CCH, 1, 0,
                                            c2_pw_w, 1, CCH, nullptr, nullptr, 0,
                                            XN, CCH, LPIX, CCH, CCH);
    dwconv<<<(LPIX * CCH + EL - 1) / EL, EL, 0, stream>>>(XN, CCH, 0, c2_dw_w, c2_dw_b,
                                                          FEA, CCH, OUTLR, CCH, 0, CCH, 0);
    upshuffle<<<(LPIX * 27 + EL - 1) / EL, EL, 0, stream>>>(OUTLR, up_w, up_b, (float*)d_out);
}
